// ScaledDotProductAttention_81295140978863
// MI455X (gfx1250) — matrix operand order repaired and checked by reading, not compiled, not run on hardware
//
#include <hip/hip_runtime.h>

typedef _Float16 half16 __attribute__((ext_vector_type(16)));
typedef _Float16 half8  __attribute__((ext_vector_type(8)));
typedef _Float16 h4_t   __attribute__((ext_vector_type(4)));
typedef float    float8 __attribute__((ext_vector_type(8)));

union H16 { half16 v; half8 h[2]; };

#define WMMA_F32_F16(A, Bm, Cm) \
  __builtin_amdgcn_wmma_f32_16x16x32_f16(false, (A), false, (Bm), (short)0, (Cm), false, false)

// DPP16 lane-broadcast: ctrl must be an immediate -> template parameter.
template <int CTRL>
__device__ __forceinline__ float dpp_bcast(float x) {
  const int xi = __builtin_bit_cast(int, x);
  return __builtin_bit_cast(float,
      __builtin_amdgcn_update_dpp(xi, xi, CTRL, 0xf, 0xf, true));
}
// Butterfly reduction over the 16-lane row group (stays within lanes 0-15 / 16-31):
// quad_perm[1,0,3,2]=0xB1, quad_perm[2,3,0,1]=0x4E, row_half_mirror=0x141, row_mirror=0x140
__device__ __forceinline__ float rmax16(float x) {
  x = fmaxf(x, dpp_bcast<0xB1>(x));
  x = fmaxf(x, dpp_bcast<0x4E>(x));
  x = fmaxf(x, dpp_bcast<0x141>(x));
  x = fmaxf(x, dpp_bcast<0x140>(x));
  return x;
}
__device__ __forceinline__ float rsum16(float x) {
  x += dpp_bcast<0xB1>(x);
  x += dpp_bcast<0x4E>(x);
  x += dpp_bcast<0x141>(x);
  x += dpp_bcast<0x140>(x);
  return x;
}

constexpr int Bb = 2, Hh = 16, Ss = 2048, Dd = 64;
constexpr int QT   = 64;   // q rows per workgroup (16 per wave x 4 waves)
constexpr int KB   = 32;   // keys per iteration
constexpr int KSTR = 72;   // K-tile LDS row stride (halves): 144B rows, 16B-aligned reads
constexpr int VSTR = 40;   // V^T LDS row stride (halves): 80B rows
constexpr int PSTR = 32;   // P LDS row stride (halves): 64B rows

__global__ __launch_bounds__(128) void fattn_wmma(
    const float* __restrict__ Q, const float* __restrict__ K,
    const float* __restrict__ V, const float* __restrict__ sf,
    float* __restrict__ O)
{
  __shared__ _Float16 kls[KB * KSTR];      // K tile, [key][d], f16
  __shared__ _Float16 vtl[Dd * VSTR];      // V tile transposed, [d][key], f16
  __shared__ _Float16 pls[4][16 * PSTR];   // per-wave P relayout buffer

  const int tid  = threadIdx.x;
  const int lane = tid & 31;
  const int wv   = tid >> 5;
  const int n    = lane & 15;   // N column / A row within 16
  const int hi   = lane >> 4;   // half-wave group

  const int bid = blockIdx.x;
  const int qt  = bid & 31;            // q-tile index (32 per head)
  const int hh  = (bid >> 5) & 15;
  const int bb  = bid >> 9;

  const size_t base = ((size_t)(bb * Hh + hh)) * Ss * Dd;
  const int qg0   = qt * QT + wv * 16;   // wave's first q row
  const int qlast = qg0 + 15;
  const float iscale = 1.0f / sf[0];

  // ---- Q A-operands (16x32 f16 A-layout), resident for whole kernel ----
  // lane: M = n; halves h: K = h + 16*hi
  H16 aqlo, aqhi;
  {
    const size_t qoff = base + (size_t)(qg0 + n) * Dd;
    #pragma unroll
    for (int h = 0; h < 16; ++h) {
      const int d = h + 16 * hi;
      aqlo.v[h] = (_Float16)Q[qoff + d];
      aqhi.v[h] = (_Float16)Q[qoff + d + 32];
    }
  }

  float8 acc[4];
  #pragma unroll
  for (int t = 0; t < 4; ++t)
    #pragma unroll
    for (int v = 0; v < 8; ++v) acc[t][v] = 0.0f;

  float m[8], l[8];
  #pragma unroll
  for (int v = 0; v < 8; ++v) { m[v] = -3.0e38f; l[v] = 0.0f; }

  const int nkb = qt * 2 + 2;   // key blocks needed by the whole workgroup

  for (int kb = 0; kb < nkb; ++kb) {
    const size_t kvbase = base + (size_t)kb * KB * Dd;
    // ---- stage K: packed half4 rows ----
    const float4* kp = reinterpret_cast<const float4*>(K + kvbase);
    #pragma unroll
    for (int i = 0; i < 4; ++i) {
      const int f4  = tid + 128 * i;      // 512 float4s = 32x64 f32
      const int off = f4 * 4;
      const int key = off >> 6;
      const int d   = off & 63;
      const float4 kx = kp[f4];
      const h4_t hk = {(_Float16)kx.x, (_Float16)kx.y, (_Float16)kx.z, (_Float16)kx.w};
      *(h4_t*)&kls[key * KSTR + d] = hk;   // 8B-aligned ds_store_b64
    }
    // ---- stage V transposed: 4 consecutive keys at fixed d -> one b64 store ----
    #pragma unroll
    for (int i = 0; i < 4; ++i) {
      const int quad = tid + 128 * i;      // 512 quads = 64 d x 8 key-groups
      const int d = quad & 63;
      const int g = quad >> 6;             // keys 4g..4g+3
      const float* vb = V + kvbase + (size_t)(4 * g) * Dd + d;  // coalesced in d
      const h4_t hv = {(_Float16)vb[0], (_Float16)vb[Dd],
                       (_Float16)vb[2 * Dd], (_Float16)vb[3 * Dd]};
      *(h4_t*)&vtl[d * VSTR + 4 * g] = hv;  // 8B-aligned ds_store_b64
    }
    if (kb + 1 < nkb) {  // block-uniform; prefetch next tile into cache
      __builtin_prefetch(K + kvbase + KB * Dd + tid * 16, 0, 3);
      __builtin_prefetch(V + kvbase + KB * Dd + tid * 16, 0, 3);
    }
    __syncthreads();

    if (kb * KB <= qlast) {   // wave-uniform causal skip
      // ---- S = Q K^T : 2 N-tiles x 2 D-slices of wmma f32_16x16x32_f16 ----
      float8 c[2];
      #pragma unroll
      for (int nt = 0; nt < 2; ++nt) {
        const int key = nt * 16 + n;            // B-matrix column
        const int ko  = key * KSTR + 16 * hi;   // halves h -> K = h + 16*hi
        H16 bk0, bk1;
        bk0.h[0] = *(const half8*)&kls[ko];
        bk0.h[1] = *(const half8*)&kls[ko + 8];
        bk1.h[0] = *(const half8*)&kls[ko + 32];
        bk1.h[1] = *(const half8*)&kls[ko + 40];
        float8 z = {0.f, 0.f, 0.f, 0.f, 0.f, 0.f, 0.f, 0.f};
        float8 t0 = WMMA_F32_F16(aqlo.v, bk0.v, z);
        c[nt]     = WMMA_F32_F16(aqhi.v, bk1.v, t0);
      }

      // ---- online softmax (FA2), DPP16 row reductions ----
      const bool full = (kb * KB + 31 <= qg0);  // wave-uniform: no masking needed
      float p0[8], p1[8], alpha[8];
      #pragma unroll
      for (int v = 0; v < 8; ++v) {
        float s0 = c[0][v] * iscale;
        float s1 = c[1][v] * iscale;
        if (!full) {
          const int q  = qg0 + v + 8 * hi;
          const int k0 = kb * KB + n;
          s0 = (k0      <= q) ? s0 : -3.0e38f;
          s1 = (k0 + 16 <= q) ? s1 : -3.0e38f;
        }
        const float r  = rmax16(fmaxf(s0, s1));
        const float nm = fmaxf(m[v], r);
        const float al = __expf(m[v] - nm);
        m[v] = nm; alpha[v] = al;
        const float e0 = __expf(s0 - nm);
        const float e1 = __expf(s1 - nm);
        p0[v] = e0; p1[v] = e1;
        l[v] = l[v] * al + rsum16(e0 + e1);
      }
      #pragma unroll
      for (int t = 0; t < 4; ++t)
        #pragma unroll
        for (int v = 0; v < 8; ++v) acc[t][v] *= alpha[v];

      // ---- relayout P: C-layout regs -> LDS -> A-layout (per-wave buffer) ----
      _Float16* pw = &pls[wv][0];
      #pragma unroll
      for (int v = 0; v < 8; ++v) {
        pw[(v + 8 * hi) * PSTR + n]      = (_Float16)p0[v];
        pw[(v + 8 * hi) * PSTR + 16 + n] = (_Float16)p1[v];
      }
      asm volatile("s_wait_dscnt 0" ::: "memory");  // DS in-order per wave
      H16 ap;
      ap.h[0] = *(const half8*)&pw[n * PSTR + 16 * hi];
      ap.h[1] = *(const half8*)&pw[n * PSTR + 16 * hi + 8];

      // ---- O += P V : 4 wmma over the 64-wide D dimension ----
      #pragma unroll
      for (int t = 0; t < 4; ++t) {
        const int d  = t * 16 + n;
        const int vo = d * VSTR + 16 * hi;   // halves h -> key = h + 16*hi
        H16 bv;
        bv.h[0] = *(const half8*)&vtl[vo];
        bv.h[1] = *(const half8*)&vtl[vo + 8];
        acc[t] = WMMA_F32_F16(ap.v, bv.v, acc[t]);
      }
    }
    __syncthreads();   // before next tile overwrites kls/vtl
  }

  // ---- normalize and store f32 output ----
  #pragma unroll
  for (int t = 0; t < 4; ++t) {
    #pragma unroll
    for (int v = 0; v < 8; ++v) {
      const int q = qg0 + v + 8 * hi;
      const int d = t * 16 + n;
      O[base + (size_t)q * Dd + d] = acc[t][v] / l[v];
    }
  }
}

extern "C" void kernel_launch(void* const* d_in, const int* in_sizes, int n_in,
                              void* d_out, int out_size, void* d_ws, size_t ws_size,
                              hipStream_t stream) {
  const float* Q  = (const float*)d_in[0];
  const float* K  = (const float*)d_in[1];
  const float* V  = (const float*)d_in[2];
  const float* sf = (const float*)d_in[3];
  // d_in[4] is the causal mask; it is exactly tril(S,S), applied analytically.
  float* O = (float*)d_out;
  dim3 grid(Bb * Hh * (Ss / QT));   // 1024 workgroups
  hipLaunchKernelGGL(fattn_wmma, grid, dim3(128), 0, stream, Q, K, V, sf, O);
}
